// MapEncoder_28956669509791
// MI455X (gfx1250) — compile-verified, matrix operand order checked by reading, not run on hardware
//
#include <hip/hip_runtime.h>
#include <cstdint>
#include <cstddef>

// MapEncoder for MI455X (gfx1250): all dense layers as bf16 WMMA GEMMs
// (v_wmma_f32_16x16x32_bf16, wave32, 16x16 tile/wave, 64-row blocks with the
// A-tile DMA'd into LDS by the Tensor Data Mover), segment-softmax attention
// via f32 atomics, LN/ReLU fused wave-per-row.

#define PI_F 3.14159265358979323846f

typedef __bf16 bf16;
typedef __attribute__((ext_vector_type(16))) __bf16 v16bf;
typedef __attribute__((ext_vector_type(8)))  __bf16 v8bf;
typedef __attribute__((ext_vector_type(8)))  float  v8f;
typedef __attribute__((ext_vector_type(4)))  unsigned int v4u;
typedef __attribute__((ext_vector_type(8)))  int v8i;
typedef __attribute__((ext_vector_type(4)))  int v4i;

// ---------------- device helpers ----------------

__device__ inline float wave_sum32(float v) {
#pragma unroll
  for (int o = 16; o > 0; o >>= 1) v += __shfl_xor(v, o, 32);
  return v;
}

__device__ inline float wrap_angle_dev(float a) {
  float r = fmodf(a + PI_F, 2.0f * PI_F);
  r = (r < 0.0f) ? r + 2.0f * PI_F : r;
  return r - PI_F;
}

// monotone float <-> uint mapping for atomicMax on floats
__device__ inline unsigned fkey(float f) {
  unsigned u = __float_as_uint(f);
  return (u >> 31) ? ~u : (u | 0x80000000u);
}
__device__ inline float funkey(unsigned u) {
  unsigned i = (u & 0x80000000u) ? (u ^ 0x80000000u) : ~u;
  return __uint_as_float(i);
}

// load a 16x32 bf16 WMMA operand fragment (per-lane 16 values).
// kofs = 0 for lanes 0-15, 8 for lanes 16-31.
__device__ inline v16bf load_frag16(const bf16* base, int kofs) {
  v8bf lo = *(const v8bf*)(base + kofs);
  v8bf hi = *(const v8bf*)(base + kofs + 16);
  v16bf r;
#pragma unroll
  for (int i = 0; i < 8; ++i) { r[i] = lo[i]; r[8 + i] = hi[i]; }
  return r;
}

// Issue a TDM 2D tile load: rows [rowBase, rowBase+BR) x K bf16 elements of a
// [M,K] row-major tensor -> LDS at ldsAddr. Rows past M are zero-filled by the
// TDM OOB rule (tensor_dim1 = M - rowBase).
template <int K, int BR>
__device__ inline void tdm_load_tile(const bf16* A, long long rowBase,
                                     long long M, unsigned ldsAddr) {
  unsigned long long ga =
      (unsigned long long)(size_t)(const void*)(A + rowBase * K);
  unsigned rowsLeft = (unsigned)(M - rowBase);
  v4u g0;
  g0[0] = 1u;                                        // count=1, user mode
  g0[1] = ldsAddr;                                   // lds_addr (bytes)
  g0[2] = (unsigned)(ga & 0xFFFFFFFFu);              // global_addr[31:0]
  g0[3] = (unsigned)((ga >> 32) & 0x01FFFFFFu) | 0x80000000u;  // addr[56:32]|type=2
  v8i g1;
  g1[0] = 0x00010000;                                // data_size=1 (2 bytes)
  g1[1] = (int)(((unsigned)K & 0xFFFFu) << 16);      // tensor_dim0[15:0]
  g1[2] = (int)((((unsigned)K) >> 16) | ((rowsLeft & 0xFFFFu) << 16));
  g1[3] = (int)((rowsLeft >> 16) | (((unsigned)K) << 16));  // dim1 hi | tile_dim0
  g1[4] = BR;                                        // tile_dim1 = BR rows
  g1[5] = (int)K;                                    // tensor_dim0_stride
  g1[6] = 0;
  g1[7] = 0;
  v4i g2 = {0, 0, 0, 0};
  v4i g3 = {0, 0, 0, 0};
#if defined(__clang_major__) && (__clang_major__ >= 23)
  v8i g4 = {0, 0, 0, 0, 0, 0, 0, 0};
  __builtin_amdgcn_tensor_load_to_lds(g0, g1, g2, g3, g4, 0);
#else
  __builtin_amdgcn_tensor_load_to_lds(g0, g1, g2, g3, 0);
#endif
  __builtin_amdgcn_s_wait_tensorcnt(0);
}

// ---------------- WMMA GEMM ----------------
// C[M,N] = A[M,K](bf16) @ Wt[N,K](bf16, pre-transposed) + bias
// block = 256 threads = 8 waves; block computes BR rows x N cols.
// BR = 64 (32 for K=512): each B fragment is reused across BR/16 WMMAs.
template <int K, int N, bool RELU, bool WF32, bool WB16>
__global__ __launch_bounds__(256)
void gemm_bf16_kernel(const bf16* __restrict__ A, const bf16* __restrict__ Wt,
                      const float* __restrict__ bias,
                      float* __restrict__ Cf, bf16* __restrict__ Cb, int M) {
  constexpr int BR = (K > 256) ? 32 : 64;
  constexpr int RT = BR / 16;
  __shared__ bf16 As[BR * K];

  const int tid = threadIdx.x;
  const int wave = tid >> 5;
  const long long rowBase = (long long)blockIdx.x * BR;

  if (wave == 0) {
    tdm_load_tile<K, BR>(A, rowBase, (long long)M,
                         (unsigned)(size_t)(const void*)As);
  }
  __syncthreads();

  const int lane = tid & 31;
  const int ln   = lane & 15;
  const int half = lane >> 4;          // 0: lanes 0-15, 1: lanes 16-31
  const int kofs = half ? 8 : 0;
  constexpr int NT = N / 16;
  const bool full = (rowBase + BR) <= (long long)M;

  for (int t = wave; t < NT; t += 8) {
    v8f z = {};
    v8f acc[RT];
#pragma unroll
    for (int i = 0; i < RT; ++i) acc[i] = z;
    const bf16* wcol = Wt + (long long)(t * 16 + ln) * K;
#pragma unroll
    for (int k0 = 0; k0 < K; k0 += 32) {
      v16bf b = load_frag16(wcol + k0, kofs);
#pragma unroll
      for (int rt = 0; rt < RT; ++rt) {
        v16bf a = load_frag16(As + (rt * 16 + ln) * K + k0, kofs);
        acc[rt] = __builtin_amdgcn_wmma_f32_16x16x32_bf16(
            false, a, false, b, (short)0, acc[rt], false, false);
      }
    }
    const int col = t * 16 + ln;
    const float bv = bias[col];
#pragma unroll
    for (int rt = 0; rt < RT; ++rt) {
#pragma unroll
      for (int r = 0; r < 8; ++r) {
        long long row = rowBase + rt * 16 + r + half * 8;
        if (full || row < (long long)M) {
          float v = acc[rt][r] + bv;
          if (RELU) v = fmaxf(v, 0.0f);
          if (WF32) Cf[row * N + col] = v;
          if (WB16) Cb[row * N + col] = (bf16)v;
        }
      }
    }
  }
}

template <int K, int N, bool RELU, bool WF32, bool WB16>
static inline void launch_gemm(const bf16* A, const bf16* Wt, const float* bias,
                               float* Cf, bf16* Cb, long long M,
                               hipStream_t stream) {
  constexpr int BR = (K > 256) ? 32 : 64;
  dim3 g((unsigned)((M + BR - 1) / BR));
  gemm_bf16_kernel<K, N, RELU, WF32, WB16>
      <<<g, dim3(256), 0, stream>>>(A, Wt, bias, Cf, Cb, (int)M);
}

// ---------------- LayerNorm (wave per 128-row) ----------------
__global__ __launch_bounds__(256)
void layernorm_kernel(const float* __restrict__ x, const float* __restrict__ g,
                      const float* __restrict__ b, float* __restrict__ outf,
                      bf16* __restrict__ outb, long long M) {
  long long row = (long long)blockIdx.x * 8 + (threadIdx.x >> 5);
  if (row >= M) return;
  int lane = threadIdx.x & 31;
  const float* xr = x + row * 128;
  float h[4], sm = 0.0f, sq = 0.0f;
#pragma unroll
  for (int i = 0; i < 4; ++i) {
    float v = xr[lane * 4 + i];
    h[i] = v; sm += v; sq += v * v;
  }
  sm = wave_sum32(sm); sq = wave_sum32(sq);
  float mean = sm * (1.0f / 128.0f);
  float inv  = rsqrtf(sq * (1.0f / 128.0f) - mean * mean + 1e-5f);
#pragma unroll
  for (int i = 0; i < 4; ++i) {
    int j = lane * 4 + i;
    float v = (h[i] - mean) * inv * g[j] + b[j];
    if (outf) outf[row * 128 + j] = v;
    if (outb) outb[row * 128 + j] = (bf16)v;
  }
}

// y = xin + LN(o; g,b); write y (f32). If g2: also write LN(y; g2,b2) as bf16.
__global__ __launch_bounds__(256)
void residual_ln_kernel(const float* __restrict__ o, const float* __restrict__ xin,
                        const float* __restrict__ g, const float* __restrict__ b,
                        const float* __restrict__ g2, const float* __restrict__ b2,
                        float* __restrict__ yout, bf16* __restrict__ ln2out,
                        long long M) {
  long long row = (long long)blockIdx.x * 8 + (threadIdx.x >> 5);
  if (row >= M) return;
  int lane = threadIdx.x & 31;
  const float* orow = o + row * 128;
  const float* xrow = xin + row * 128;
  float h[4], sm = 0.0f, sq = 0.0f;
#pragma unroll
  for (int i = 0; i < 4; ++i) {
    float v = orow[lane * 4 + i];
    h[i] = v; sm += v; sq += v * v;
  }
  sm = wave_sum32(sm); sq = wave_sum32(sq);
  float mean = sm * (1.0f / 128.0f);
  float inv  = rsqrtf(sq * (1.0f / 128.0f) - mean * mean + 1e-5f);
  float y[4]; float sm2 = 0.0f, sq2 = 0.0f;
#pragma unroll
  for (int i = 0; i < 4; ++i) {
    int j = lane * 4 + i;
    float v = xrow[j] + (h[i] - mean) * inv * g[j] + b[j];
    y[i] = v; sm2 += v; sq2 += v * v;
    yout[row * 128 + j] = v;
  }
  if (g2) {
    sm2 = wave_sum32(sm2); sq2 = wave_sum32(sq2);
    float mean2 = sm2 * (1.0f / 128.0f);
    float inv2  = rsqrtf(sq2 * (1.0f / 128.0f) - mean2 * mean2 + 1e-5f);
#pragma unroll
    for (int i = 0; i < 4; ++i) {
      int j = lane * 4 + i;
      ln2out[row * 128 + j] = (bf16)((y[i] - mean2) * inv2 * g2[j] + b2[j]);
    }
  }
}

// ---------------- first MLP layers (fused feature+LN+ReLU -> bf16) ----------
__global__ __launch_bounds__(256)
void node_mlp1_kernel(const float* __restrict__ len, const float* __restrict__ W1,
                      const float* __restrict__ b1, const float* __restrict__ g1,
                      const float* __restrict__ bt1, bf16* __restrict__ out,
                      long long M) {
  long long node = (long long)blockIdx.x * 8 + (threadIdx.x >> 5);
  if (node >= M) return;
  int lane = threadIdx.x & 31;
  float x = len[node];
  float h[4], sm = 0.0f, sq = 0.0f;
#pragma unroll
  for (int i = 0; i < 4; ++i) {
    int j = lane * 4 + i;
    float v = x * W1[j] + b1[j];
    h[i] = v; sm += v; sq += v * v;
  }
  sm = wave_sum32(sm); sq = wave_sum32(sq);
  float mean = sm * (1.0f / 128.0f);
  float inv  = rsqrtf(sq * (1.0f / 128.0f) - mean * mean + 1e-5f);
#pragma unroll
  for (int i = 0; i < 4; ++i) {
    int j = lane * 4 + i;
    float v = (h[i] - mean) * inv * g1[j] + bt1[j];
    out[node * 128 + j] = (bf16)fmaxf(v, 0.0f);
  }
}

__global__ __launch_bounds__(256)
void edge_mlp1_p2l_kernel(const float* __restrict__ p_pos, const float* __restrict__ p_head,
                          const float* __restrict__ p_side, const float* __restrict__ l_pos,
                          const float* __restrict__ l_head, const int* __restrict__ src,
                          const int* __restrict__ dst, const float* __restrict__ W1,
                          const float* __restrict__ b1, const float* __restrict__ g1,
                          const float* __restrict__ bt1, bf16* __restrict__ out,
                          long long E) {
  long long e = (long long)blockIdx.x * 8 + (threadIdx.x >> 5);
  if (e >= E) return;
  int lane = threadIdx.x & 31;
  int s = src[e], d = dst[e];
  float dx = p_pos[2 * s] - l_pos[2 * d];
  float dy = p_pos[2 * s + 1] - l_pos[2 * d + 1];
  float hd = l_head[d];
  float c = cosf(hd), sn = sinf(hd);
  float lx = c * dx + sn * dy, ly = -sn * dx + c * dy;
  float f0 = sqrtf(lx * lx + ly * ly);
  float f1 = atan2f(ly, lx);
  float f2 = wrap_angle_dev(p_head[s] - hd);
  float f3 = p_side[s];
  float h[4], sm = 0.0f, sq = 0.0f;
#pragma unroll
  for (int i = 0; i < 4; ++i) {
    int j = lane * 4 + i;
    float v = f0 * W1[j] + f1 * W1[128 + j] + f2 * W1[256 + j] + f3 * W1[384 + j] + b1[j];
    h[i] = v; sm += v; sq += v * v;
  }
  sm = wave_sum32(sm); sq = wave_sum32(sq);
  float mean = sm * (1.0f / 128.0f);
  float inv  = rsqrtf(sq * (1.0f / 128.0f) - mean * mean + 1e-5f);
#pragma unroll
  for (int i = 0; i < 4; ++i) {
    int j = lane * 4 + i;
    float v = (h[i] - mean) * inv * g1[j] + bt1[j];
    out[e * 128 + j] = (bf16)fmaxf(v, 0.0f);
  }
}

__global__ __launch_bounds__(256)
void edge_mlp1_l2l_kernel(const float* __restrict__ l_pos, const float* __restrict__ l_head,
                          const int* __restrict__ src, const int* __restrict__ dst,
                          const int* __restrict__ type_idx, const int* __restrict__ hop,
                          const float* __restrict__ W1, const float* __restrict__ b1,
                          const float* __restrict__ g1, const float* __restrict__ bt1,
                          bf16* __restrict__ out, long long E) {
  long long e = (long long)blockIdx.x * 8 + (threadIdx.x >> 5);
  if (e >= E) return;
  int lane = threadIdx.x & 31;
  int s = src[e], d = dst[e];
  float dx = l_pos[2 * s] - l_pos[2 * d];
  float dy = l_pos[2 * s + 1] - l_pos[2 * d + 1];
  float hd = l_head[d];
  float c = cosf(hd), sn = sinf(hd);
  float lx = c * dx + sn * dy, ly = -sn * dx + c * dy;
  float f[8];
  f[0] = sqrtf(lx * lx + ly * ly);
  f[1] = atan2f(ly, lx);
  f[2] = wrap_angle_dev(l_head[s] - hd);
  f[3] = (float)hop[e];
  int ty = type_idx[e];
  f[4] = (ty == 0) ? 1.0f : 0.0f;
  f[5] = (ty == 1) ? 1.0f : 0.0f;
  f[6] = (ty == 2) ? 1.0f : 0.0f;
  f[7] = (ty == 3) ? 1.0f : 0.0f;
  float h[4], sm = 0.0f, sq = 0.0f;
#pragma unroll
  for (int i = 0; i < 4; ++i) {
    int j = lane * 4 + i;
    float v = b1[j];
#pragma unroll
    for (int ff = 0; ff < 8; ++ff) v += f[ff] * W1[ff * 128 + j];
    h[i] = v; sm += v; sq += v * v;
  }
  sm = wave_sum32(sm); sq = wave_sum32(sq);
  float mean = sm * (1.0f / 128.0f);
  float inv  = rsqrtf(sq * (1.0f / 128.0f) - mean * mean + 1e-5f);
#pragma unroll
  for (int i = 0; i < 4; ++i) {
    int j = lane * 4 + i;
    float v = (h[i] - mean) * inv * g1[j] + bt1[j];
    out[e * 128 + j] = (bf16)fmaxf(v, 0.0f);
  }
}

// ---------------- segment softmax attention ----------------
__global__ __launch_bounds__(256)
void edge_sim_kernel(const float* __restrict__ q, const float* __restrict__ k,
                     const float* __restrict__ ke, const int* __restrict__ src,
                     const int* __restrict__ dst, float* __restrict__ sim,
                     unsigned* __restrict__ mmax, long long E8) {
  long long idx = (long long)blockIdx.x * blockDim.x + threadIdx.x;
  if (idx >= E8) return;
  long long e = idx >> 3;
  int h = (int)(idx & 7);
  long long s = src[e], d = dst[e];
  float acc = 0.0f;
#pragma unroll
  for (int t = 0; t < 16; ++t) {
    int c = h * 16 + t;
    acc += q[d * 128 + c] * (k[s * 128 + c] + ke[e * 128 + c]);
  }
  acc *= 0.25f;  // HD^-0.5, HD = 16
  sim[idx] = acc;
  atomicMax(&mmax[d * 8 + h], fkey(acc));
}

__global__ __launch_bounds__(256)
void edge_exp_kernel(float* __restrict__ sim, const unsigned* __restrict__ mmax,
                     const int* __restrict__ dst, float* __restrict__ den,
                     long long E8) {
  long long idx = (long long)blockIdx.x * blockDim.x + threadIdx.x;
  if (idx >= E8) return;
  long long e = idx >> 3;
  int h = (int)(idx & 7);
  long long d = dst[e];
  float m = funkey(mmax[d * 8 + h]);
  float ex = __expf(sim[idx] - m);
  sim[idx] = ex;
  atomicAdd(&den[d * 8 + h], ex);
}

__global__ __launch_bounds__(256)
void edge_agg_kernel(const float* __restrict__ ex, const float* __restrict__ den,
                     const float* __restrict__ v, const float* __restrict__ ve,
                     const int* __restrict__ src, const int* __restrict__ dst,
                     float* __restrict__ agg, long long E) {
  long long e = (long long)blockIdx.x * 8 + (threadIdx.x >> 5);
  if (e >= E) return;
  int lane = threadIdx.x & 31;
  int j0 = lane * 4;
  int h = j0 >> 4;
  long long s = src[e], d = dst[e];
  float a = ex[e * 8 + h] / den[d * 8 + h];
#pragma unroll
  for (int i = 0; i < 4; ++i) {
    int j = j0 + i;
    atomicAdd(&agg[d * 128 + j], a * (v[s * 128 + j] + ve[e * 128 + j]));
  }
}

// ---------------- gating / small elementwise ----------------
__global__ __launch_bounds__(256)
void concat_kernel(const float* __restrict__ agg, const bf16* __restrict__ xd,
                   bf16* __restrict__ out, long long M) {
  long long i = (long long)blockIdx.x * blockDim.x + threadIdx.x;
  if (i >= M * 256) return;
  long long row = i >> 8;
  int c = (int)(i & 255);
  out[i] = (c < 128) ? (bf16)agg[row * 128 + c] : (bf16)xd[row * 128 + (c - 128)];
}

__global__ __launch_bounds__(256)
void gate_update_kernel(const float* __restrict__ glin, const float* __restrict__ s,
                        const float* __restrict__ agg, bf16* __restrict__ out,
                        long long n) {
  long long i = (long long)blockIdx.x * blockDim.x + threadIdx.x;
  if (i >= n) return;
  float g = 1.0f / (1.0f + __expf(-glin[i]));
  float a = agg[i];
  out[i] = (bf16)(a + g * (s[i] - a));
}

__global__ __launch_bounds__(256)
void fill_u32_kernel(unsigned* __restrict__ p, unsigned v, long long n) {
  long long i = (long long)blockIdx.x * blockDim.x + threadIdx.x;
  if (i < n) p[i] = v;
}

// W[K,N] f32 -> Wt[N,K] bf16
__global__ __launch_bounds__(256)
void w_to_bf16t_kernel(const float* __restrict__ W, bf16* __restrict__ Wt,
                       int K, int N) {
  long long i = (long long)blockIdx.x * blockDim.x + threadIdx.x;
  if (i >= (long long)K * N) return;
  int k = (int)(i / N), n = (int)(i % N);
  Wt[(long long)n * K + k] = (bf16)W[i];
}

// ---------------- host ----------------

struct AttnWt { bf16 *wq, *wk, *wv, *wke, *wve, *ws, *wo, *wg, *wf1, *wf2; };

extern "C" void kernel_launch(void* const* d_in, const int* in_sizes, int n_in,
                              void* d_out, int out_size, void* d_ws, size_t ws_size,
                              hipStream_t stream) {
  (void)n_in; (void)out_size; (void)ws_size;
  const long long P  = in_sizes[0];
  const long long L  = in_sizes[4];
  const long long EP = in_sizes[7];
  const long long EL = in_sizes[9];

  auto F = [&](int i) { return (const float*)d_in[i]; };
  auto I = [&](int i) { return (const int*)d_in[i]; };

  const float* p_length   = F(0);
  const float* p_position = F(1);
  const float* p_heading  = F(2);
  const float* p_side     = F(3);
  const float* l_length   = F(4);
  const float* l_position = F(5);
  const float* l_heading  = F(6);
  const int* p2l_src = I(7);
  const int* p2l_dst = I(8);
  const int* l2l_src = I(9);
  const int* l2l_dst = I(10);
  const int* l2l_ty  = I(11);
  const int* l2l_hop = I(12);

  // param leaves in setup_inputs insertion order:
  // mlp: W1,b1,g1,bt1,W2,b2 ; attn: Wq,bq,Wk,bk,Wv,bv,Wke,bke,Wve,bve,Ws,bs,
  //  Wo,bo,Wg,bg,Wf1,bf1,Wf2,bf2, ln_src_g/b, ln_dst_g/b, ln_e_g/b,
  //  ln_post_g/b, ln_ffpre_g/b, ln_ffpost_g/b
  const int PB = 13;
  const int MB_P = PB + 0, MB_L = PB + 6, MB_P2L = PB + 12, MB_L2L = PB + 18;
  const int AB1 = PB + 24, AB2 = PB + 56;

  size_t off = 0;
  auto alloc = [&](size_t bytes) -> void* {
    off = (off + 255) & ~(size_t)255;
    void* p = (char*)d_ws + off;
    off += bytes;
    return p;
  };

  // bf16 transposed weights
  bf16* wt_pemb2 = (bf16*)alloc(128 * 128 * 2);
  bf16* wt_lemb2 = (bf16*)alloc(128 * 128 * 2);
  bf16* wt_p2l2  = (bf16*)alloc(128 * 128 * 2);
  bf16* wt_l2l2  = (bf16*)alloc(128 * 128 * 2);
  AttnWt AW[2];
  for (int a = 0; a < 2; ++a) {
    AW[a].wq  = (bf16*)alloc(128 * 128 * 2);
    AW[a].wk  = (bf16*)alloc(128 * 128 * 2);
    AW[a].wv  = (bf16*)alloc(128 * 128 * 2);
    AW[a].wke = (bf16*)alloc(128 * 128 * 2);
    AW[a].wve = (bf16*)alloc(128 * 128 * 2);
    AW[a].ws  = (bf16*)alloc(128 * 128 * 2);
    AW[a].wo  = (bf16*)alloc(128 * 128 * 2);
    AW[a].wg  = (bf16*)alloc(256 * 128 * 2);
    AW[a].wf1 = (bf16*)alloc(128 * 512 * 2);
    AW[a].wf2 = (bf16*)alloc(512 * 128 * 2);
  }

  const long long maxN = (P > L) ? P : L;
  const long long maxE = (EL > EP) ? EL : EP;

  bf16*  nodeA   = (bf16*)alloc((size_t)maxN * 128 * 2);
  float* p_embs  = (float*)alloc((size_t)P * 128 * 4);
  float* l_embs  = (float*)alloc((size_t)L * 128 * 4);
  float* l_embs2 = (float*)alloc((size_t)L * 128 * 4);
  bf16*  edgeA   = (bf16*)alloc((size_t)maxE * 128 * 2);  // also used as e_ln
  float* attr    = (float*)alloc((size_t)maxE * 128 * 4);
  float* kebuf   = (float*)alloc((size_t)maxE * 128 * 4);
  float* vebuf   = (float*)alloc((size_t)maxE * 128 * 4);
  float* simbuf  = (float*)alloc((size_t)maxE * 8 * 4);
  unsigned* mbuf = (unsigned*)alloc((size_t)L * 8 * 4);
  float* denbuf  = (float*)alloc((size_t)L * 8 * 4);
  float* qbuf    = (float*)alloc((size_t)L * 128 * 4);
  float* kbuf    = (float*)alloc((size_t)P * 128 * 4);
  float* vbuf    = (float*)alloc((size_t)P * 128 * 4);
  bf16*  src_ln  = (bf16*)alloc((size_t)P * 128 * 2);
  bf16*  dst_ln  = (bf16*)alloc((size_t)L * 128 * 2);
  float* aggbuf  = (float*)alloc((size_t)L * 128 * 4);
  bf16*  catbuf  = (bf16*)alloc((size_t)L * 256 * 2);
  float* glin    = (float*)alloc((size_t)L * 128 * 4);
  float* sbuf    = (float*)alloc((size_t)L * 128 * 4);
  bf16*  agg2    = (bf16*)alloc((size_t)L * 128 * 2);
  float* obuf    = (float*)alloc((size_t)L * 128 * 4);
  float* xnew    = (float*)alloc((size_t)L * 128 * 4);
  bf16*  ffpre   = (bf16*)alloc((size_t)L * 128 * 2);
  bf16*  h1      = (bf16*)alloc((size_t)L * 512 * 2);
  float* h2      = (float*)alloc((size_t)L * 128 * 4);

  const dim3 blk(256);
  auto rows = [](long long n) { return dim3((unsigned)((n + 7) / 8)); };
  auto flat = [](long long n) { return dim3((unsigned)((n + 255) / 256)); };

  auto convW = [&](const float* W, bf16* Wt, int K, int N) {
    long long n = (long long)K * N;
    w_to_bf16t_kernel<<<flat(n), blk, 0, stream>>>(W, Wt, K, N);
  };

  // ---- convert weights
  convW(F(MB_P   + 4), wt_pemb2, 128, 128);
  convW(F(MB_L   + 4), wt_lemb2, 128, 128);
  convW(F(MB_P2L + 4), wt_p2l2,  128, 128);
  convW(F(MB_L2L + 4), wt_l2l2,  128, 128);
  const int ABs[2] = {AB1, AB2};
  for (int a = 0; a < 2; ++a) {
    int AB = ABs[a];
    convW(F(AB + 0),  AW[a].wq,  128, 128);
    convW(F(AB + 2),  AW[a].wk,  128, 128);
    convW(F(AB + 4),  AW[a].wv,  128, 128);
    convW(F(AB + 6),  AW[a].wke, 128, 128);
    convW(F(AB + 8),  AW[a].wve, 128, 128);
    convW(F(AB + 10), AW[a].ws,  128, 128);
    convW(F(AB + 12), AW[a].wo,  128, 128);
    convW(F(AB + 14), AW[a].wg,  256, 128);
    convW(F(AB + 16), AW[a].wf1, 128, 512);
    convW(F(AB + 18), AW[a].wf2, 512, 128);
  }

  // ---- node embeddings
  node_mlp1_kernel<<<rows(P), blk, 0, stream>>>(
      p_length, F(MB_P + 0), F(MB_P + 1), F(MB_P + 2), F(MB_P + 3), nodeA, P);
  launch_gemm<128, 128, false, true, false>(nodeA, wt_pemb2, F(MB_P + 5),
                                            p_embs, nullptr, P, stream);
  node_mlp1_kernel<<<rows(L), blk, 0, stream>>>(
      l_length, F(MB_L + 0), F(MB_L + 1), F(MB_L + 2), F(MB_L + 3), nodeA, L);
  launch_gemm<128, 128, false, true, false>(nodeA, wt_lemb2, F(MB_L + 5),
                                            l_embs, nullptr, L, stream);

  // ---- attention runner
  auto run_attn = [&](int AB, const AttnWt& W, const float* x_src, long long Ns,
                      const float* x_dst_in, long long Nd, const int* src,
                      const int* dst, long long E, const float* eattr, bool self,
                      float* out) {
    bf16* e_ln = edgeA;  // first-layer edge activations are dead here; reuse
    layernorm_kernel<<<rows(Ns), blk, 0, stream>>>(x_src, F(AB + 20), F(AB + 21),
                                                   nullptr, src_ln, Ns);
    const bf16* dln = src_ln;
    if (!self) {
      layernorm_kernel<<<rows(Nd), blk, 0, stream>>>(x_dst_in, F(AB + 22),
                                                     F(AB + 23), nullptr, dst_ln, Nd);
      dln = dst_ln;
    }
    layernorm_kernel<<<rows(E), blk, 0, stream>>>(eattr, F(AB + 24), F(AB + 25),
                                                  nullptr, e_ln, E);
    launch_gemm<128, 128, false, true, false>(dln, W.wq, F(AB + 1), qbuf, nullptr, Nd, stream);
    launch_gemm<128, 128, false, true, false>(src_ln, W.wk, F(AB + 3), kbuf, nullptr, Ns, stream);
    launch_gemm<128, 128, false, true, false>(src_ln, W.wv, F(AB + 5), vbuf, nullptr, Ns, stream);
    launch_gemm<128, 128, false, true, false>(e_ln, W.wke, F(AB + 7), kebuf, nullptr, E, stream);
    launch_gemm<128, 128, false, true, false>(e_ln, W.wve, F(AB + 9), vebuf, nullptr, E, stream);

    fill_u32_kernel<<<flat(Nd * 8), blk, 0, stream>>>(mbuf, 0u, Nd * 8);
    fill_u32_kernel<<<flat(Nd * 8), blk, 0, stream>>>((unsigned*)denbuf, 0u, Nd * 8);
    fill_u32_kernel<<<flat(Nd * 128), blk, 0, stream>>>((unsigned*)aggbuf, 0u, Nd * 128);

    edge_sim_kernel<<<flat(E * 8), blk, 0, stream>>>(qbuf, kbuf, kebuf, src, dst,
                                                     simbuf, mbuf, E * 8);
    edge_exp_kernel<<<flat(E * 8), blk, 0, stream>>>(simbuf, mbuf, dst, denbuf, E * 8);
    edge_agg_kernel<<<rows(E), blk, 0, stream>>>(simbuf, denbuf, vbuf, vebuf, src,
                                                 dst, aggbuf, E);

    concat_kernel<<<flat(Nd * 256), blk, 0, stream>>>(aggbuf, dln, catbuf, Nd);
    launch_gemm<256, 128, false, true, false>(catbuf, W.wg, F(AB + 15), glin, nullptr, Nd, stream);
    launch_gemm<128, 128, false, true, false>(dln, W.ws, F(AB + 11), sbuf, nullptr, Nd, stream);
    gate_update_kernel<<<flat(Nd * 128), blk, 0, stream>>>(glin, sbuf, aggbuf, agg2,
                                                           Nd * 128);
    launch_gemm<128, 128, false, true, false>(agg2, W.wo, F(AB + 13), obuf, nullptr, Nd, stream);
    residual_ln_kernel<<<rows(Nd), blk, 0, stream>>>(
        obuf, x_dst_in, F(AB + 26), F(AB + 27), F(AB + 28), F(AB + 29), xnew, ffpre, Nd);
    launch_gemm<128, 512, true, false, true>(ffpre, W.wf1, F(AB + 17), nullptr, h1, Nd, stream);
    launch_gemm<512, 128, false, true, false>(h1, W.wf2, F(AB + 19), h2, nullptr, Nd, stream);
    residual_ln_kernel<<<rows(Nd), blk, 0, stream>>>(
        h2, xnew, F(AB + 30), F(AB + 31), nullptr, nullptr, out, nullptr, Nd);
  };

  // ---- p2l edge attributes + attention 1 (polyline -> lane)
  edge_mlp1_p2l_kernel<<<rows(EP), blk, 0, stream>>>(
      p_position, p_heading, p_side, l_position, l_heading, p2l_src, p2l_dst,
      F(MB_P2L + 0), F(MB_P2L + 1), F(MB_P2L + 2), F(MB_P2L + 3), edgeA, EP);
  launch_gemm<128, 128, false, true, false>(edgeA, wt_p2l2, F(MB_P2L + 5),
                                            attr, nullptr, EP, stream);
  run_attn(AB1, AW[0], p_embs, P, l_embs, L, p2l_src, p2l_dst, EP, attr, false,
           l_embs2);

  // ---- l2l edge attributes + attention 2 (lane self-attention)
  edge_mlp1_l2l_kernel<<<rows(EL), blk, 0, stream>>>(
      l_position, l_heading, l2l_src, l2l_dst, l2l_ty, l2l_hop,
      F(MB_L2L + 0), F(MB_L2L + 1), F(MB_L2L + 2), F(MB_L2L + 3), edgeA, EL);
  launch_gemm<128, 128, false, true, false>(edgeA, wt_l2l2, F(MB_L2L + 5),
                                            attr, nullptr, EL, stream);
  run_attn(AB2, AW[1], l_embs2, L, l_embs2, L, l2l_src, l2l_dst, EL, attr, true,
           (float*)d_out);
}
